// POS_Inject_60009283060003
// MI455X (gfx1250) — matrix operand order repaired and checked by reading, not compiled, not run on hardware
//
#include <hip/hip_runtime.h>

typedef __attribute__((ext_vector_type(16))) _Float16 v16h;
typedef __attribute__((ext_vector_type(8)))  float    v8f;

union Frag {
  v16h h;
  unsigned int u[8];
};

__device__ __forceinline__ v8f wmma_f16(v16h a, v16h b, v8f c) {
  return __builtin_amdgcn_wmma_f32_16x16x32_f16(false, a, false, b, (short)0, c,
                                                false, false);
}

// K-pair start offsets per CDNA5 ISA VGPR layouts (16-bit operands, wave32).
// A 16x32: lanes 0-15 VGPR0-3 K=0..7, VGPR4-7 K=16..23; lanes 16-31 are +8.
__device__ __forceinline__ int ksA(int j, int hi) {
  return 2 * j + ((j & 4) ? 8 : 0) + 8 * hi;
}
// B 32x16: lanes 0-15 VGPR0-3 K=0..7, VGPR4-7 K=16..23; lanes 16-31 are +8.
__device__ __forceinline__ int ksB(int j, int hi) { return 2 * j + ((j & 4) ? 8 : 0) + 8 * hi; }

// ---------------------------------------------------------------------------
// 1) xbar(b,c,h) = mean_w x(b,c,h,w).  One wave per row of 64 floats.
__global__ void k_rowmean(const float* __restrict__ x, float* __restrict__ xbar) {
  int r = blockIdx.x * 8 + (threadIdx.x >> 5);   // (b*512+c)*64+h
  int l = threadIdx.x & 31;
  float2 v = *(const float2*)(x + (size_t)r * 64 + l * 2);
  float s = v.x + v.y;
  for (int off = 16; off > 0; off >>= 1) s += __shfl_xor(s, off, 32);
  if (l == 0) xbar[r] = s * (1.0f / 64.0f);
}

// ---------------------------------------------------------------------------
// 2) x1d = pool4(xbar) + pos_emb[ph]; up = linear upsample 16 -> 64.
__global__ void k_upsample(const float* __restrict__ xbar,
                           const int* __restrict__ pos_h,
                           const float* __restrict__ pos_emb,
                           float* __restrict__ up) {
  int bc = blockIdx.x;          // b*512 + c
  int b = bc >> 9, c = bc & 511;
  int t = threadIdx.x;          // 0..63
  __shared__ float x1d[16];
  if (t < 16) {
    const float* xb = xbar + (size_t)bc * 64 + t * 4;
    float s = 0.25f * (xb[0] + xb[1] + xb[2] + xb[3]);
    int ph = pos_h[(size_t)b * 4096 + (t * 4) * 64] >> 3;   // //POS_RFACTOR
    s += pos_emb[ph * 512 + c];
    x1d[t] = s;
  }
  __syncthreads();
  float src = (float)t * 15.0f / 63.0f;
  int lo = (int)floorf(src);
  int hi = min(lo + 1, 15);
  float f = src - (float)lo;
  up[(size_t)bc * 64 + t] = x1d[lo] * (1.0f - f) + x1d[hi] * f;
}

// ---------------------------------------------------------------------------
// 3) k1(b,h,o) = Wk(o,:) . up(b,:,h) + bk(o)
__global__ void k_k1(const float* __restrict__ up, const float* __restrict__ Wk,
                     const float* __restrict__ bk, float* __restrict__ k1) {
  int b = blockIdx.x >> 6, h = blockIdx.x & 63;
  int o = threadIdx.x;          // 0..255
  __shared__ float ups[512];
  ups[o]       = up[((size_t)(b * 512 + o)) * 64 + h];
  ups[o + 256] = up[((size_t)(b * 512 + o + 256)) * 64 + h];
  __syncthreads();
  float acc = bk[o];
  const float* w = Wk + (size_t)o * 512;
  for (int c = 0; c < 512; ++c) acc += w[c] * ups[c];
  k1[((size_t)(b * 64 + h)) * 256 + o] = acc;
}

// ---------------------------------------------------------------------------
// 4) Kt(b,h,c) = sum_o Wq(o,c) k1(b,h,o)  (f16);  e0(b,h) = bq . k1(b,h,:)
__global__ void k_kt(const float* __restrict__ k1, const float* __restrict__ Wq,
                     const float* __restrict__ bq, _Float16* __restrict__ Kt16,
                     float* __restrict__ e0) {
  int b = blockIdx.x >> 6, h = blockIdx.x & 63;
  int t = threadIdx.x;          // 0..255
  __shared__ float k1s[256];
  __shared__ float red[256];
  k1s[t] = k1[((size_t)(b * 64 + h)) * 256 + t];
  __syncthreads();
  for (int cc = t; cc < 512; cc += 256) {
    float acc = 0.f;
    for (int o = 0; o < 256; ++o) acc += Wq[(size_t)o * 512 + cc] * k1s[o];
    Kt16[((size_t)(b * 64 + h)) * 512 + cc] = (_Float16)acc;
  }
  red[t] = bq[t] * k1s[t];
  __syncthreads();
  for (int s = 128; s > 0; s >>= 1) {
    if (t < s) red[t] += red[t + s];
    __syncthreads();
  }
  if (t == 0) e0[b * 64 + h] = red[0];
}

// ---------------------------------------------------------------------------
// 5) Vmean(b,o,h) = Wv(o,:) . xbar(b,:,h) + bv(o)   (f16, h contiguous)
__global__ void k_vmean(const float* __restrict__ xbar, const float* __restrict__ Wv,
                        const float* __restrict__ bv, _Float16* __restrict__ Vm16) {
  int b = blockIdx.x >> 9, o = blockIdx.x & 511;
  int h = threadIdx.x;          // 0..63
  float acc = bv[o];
  const float* w = Wv + (size_t)o * 512;
  const float* xb = xbar + (size_t)b * 512 * 64 + h;
  for (int c = 0; c < 512; ++c) acc += w[c] * xb[c * 64];
  Vm16[((size_t)(b * 512 + o)) * 64 + h] = (_Float16)acc;
}

// ---------------------------------------------------------------------------
// 6) E(b,n,h) = x(b,:,n)^T Kt(b,:,h) + e0(b,h).  WMMA f16, M=n N=h K=c.
//    Per block: 128(n) x 64(h) tile, 8 waves each own a 16-row strip.
__global__ __launch_bounds__(256) void k_energy(const float* __restrict__ x,
                                                const _Float16* __restrict__ Kt16,
                                                const float* __restrict__ e0,
                                                float* __restrict__ E) {
  const int b = blockIdx.x >> 5;
  const int n0 = (blockIdx.x & 31) * 128;
  const int t = threadIdx.x;
  const int w = t >> 5, l = t & 31;
  const int lm = l & 15, hi = l >> 4;
  __shared__ _Float16 As[128 * 36];   // [m][k], row stride 36 halves
  __shared__ _Float16 Bs[64 * 36];    // [h][k]
  v8f acc[4] = {};
  for (int kk = 0; kk < 512; kk += 32) {
    {  // stage A: 32(c) x 128(n) f32 -> f16
      int kc = t >> 3, seg = t & 7;
      const float* srcf = x + ((size_t)(b * 512 + kk + kc)) * 4096 + n0 + seg * 16;
      if (kk + 32 < 512) __builtin_prefetch(srcf + (size_t)32 * 4096, 0, 1);
      float4 va = ((const float4*)srcf)[0];
      float4 vb = ((const float4*)srcf)[1];
      float4 vc = ((const float4*)srcf)[2];
      float4 vd = ((const float4*)srcf)[3];
      float tmp[16] = {va.x, va.y, va.z, va.w, vb.x, vb.y, vb.z, vb.w,
                       vc.x, vc.y, vc.z, vc.w, vd.x, vd.y, vd.z, vd.w};
      int m0 = seg * 16;
#pragma unroll
      for (int i = 0; i < 16; ++i) As[(m0 + i) * 36 + kc] = (_Float16)tmp[i];
    }
    {  // stage B: 64(h) x 32(c) f16
      int hr = t >> 2, ko = (t & 3) * 8;
      const _Float16* kb = Kt16 + ((size_t)(b * 64 + hr)) * 512 + kk + ko;
#pragma unroll
      for (int i = 0; i < 8; ++i) Bs[hr * 36 + ko + i] = kb[i];
    }
    __syncthreads();
    Frag a;
#pragma unroll
    for (int j = 0; j < 8; ++j)
      a.u[j] = *(const unsigned int*)(As + (w * 16 + lm) * 36 + ksA(j, hi));
#pragma unroll
    for (int ht = 0; ht < 4; ++ht) {
      Frag bf;
#pragma unroll
      for (int j = 0; j < 8; ++j)
        bf.u[j] = *(const unsigned int*)(Bs + (ht * 16 + lm) * 36 + ksB(j, hi));
      acc[ht] = wmma_f16(a.h, bf.h, acc[ht]);
    }
    __syncthreads();
  }
#pragma unroll
  for (int ht = 0; ht < 4; ++ht) {
    int hcol = ht * 16 + lm;
    float e = e0[b * 64 + hcol];
#pragma unroll
    for (int r = 0; r < 8; ++r) {
      int ng = n0 + w * 16 + r + 8 * hi;
      E[((size_t)(b * 4096 + ng)) * 64 + hcol] = acc[ht][r] + e;
    }
  }
}

// ---------------------------------------------------------------------------
// 7) p(b,n,h) = softmax_h E(b,n,h)  -> f16.  One wave per row of 64.
__global__ void k_softmax(const float* __restrict__ E, _Float16* __restrict__ p16) {
  int row = blockIdx.x * 8 + (threadIdx.x >> 5);
  int l = threadIdx.x & 31;
  const float* e = E + (size_t)row * 64;
  float v0 = e[l], v1 = e[l + 32];
  float m = fmaxf(v0, v1);
  for (int off = 16; off > 0; off >>= 1) m = fmaxf(m, __shfl_xor(m, off, 32));
  float a0 = __expf(v0 - m), a1 = __expf(v1 - m);
  float s = a0 + a1;
  for (int off = 16; off > 0; off >>= 1) s += __shfl_xor(s, off, 32);
  float inv = 1.0f / s;
  p16[(size_t)row * 64 + l] = (_Float16)(a0 * inv);
  p16[(size_t)row * 64 + l + 32] = (_Float16)(a1 * inv);
}

// ---------------------------------------------------------------------------
// 8) out(b,c,n) = gamma * sum_h Vmean(b,c,h) p(b,n,h) + x(b,c,n).
//    WMMA: M=c N=n K=h(64).  Block tile 128(c) x 64(n).
__global__ __launch_bounds__(256) void k_attnout(const _Float16* __restrict__ Vm16,
                                                 const _Float16* __restrict__ p16,
                                                 const float* __restrict__ x,
                                                 const float* __restrict__ gamma,
                                                 float* __restrict__ out) {
  const int b = blockIdx.x >> 8;
  const int c0 = ((blockIdx.x >> 6) & 3) * 128;
  const int n0 = (blockIdx.x & 63) * 64;
  const int t = threadIdx.x;
  const int w = t >> 5, l = t & 31;
  const int lm = l & 15, hi = l >> 4;
  __shared__ _Float16 Vs[128 * 72];   // [c][h], stride 72 halves
  __shared__ _Float16 Ps[64 * 72];    // [n][h]
  {  // stage Vmean slice: 128 x 64 halves
    int c = t >> 1, half = t & 1;
    const uint4* src =
        (const uint4*)(Vm16 + ((size_t)(b * 512 + c0 + c)) * 64 + half * 32);
    uint4* dst = (uint4*)(Vs + c * 72 + half * 32);
    dst[0] = src[0]; dst[1] = src[1]; dst[2] = src[2]; dst[3] = src[3];
  }
  {  // stage p slice: 64 x 64 halves
    int n = t >> 2, seg = t & 3;
    const uint4* src =
        (const uint4*)(p16 + ((size_t)(b * 4096 + n0 + n)) * 64 + seg * 16);
    uint4* dst = (uint4*)(Ps + n * 72 + seg * 16);
    dst[0] = src[0]; dst[1] = src[1];
  }
  __syncthreads();
  v8f acc[4] = {};
#pragma unroll
  for (int ks = 0; ks < 2; ++ks) {
    int kb = ks * 32;
    Frag a;
#pragma unroll
    for (int j = 0; j < 8; ++j)
      a.u[j] = *(const unsigned int*)(Vs + (w * 16 + lm) * 72 + kb + ksA(j, hi));
#pragma unroll
    for (int ht = 0; ht < 4; ++ht) {
      Frag bf;
#pragma unroll
      for (int j = 0; j < 8; ++j)
        bf.u[j] = *(const unsigned int*)(Ps + (ht * 16 + lm) * 72 + kb + ksB(j, hi));
      acc[ht] = wmma_f16(a.h, bf.h, acc[ht]);
    }
  }
  float g = gamma[0];
#pragma unroll
  for (int ht = 0; ht < 4; ++ht) {
    int ng = n0 + ht * 16 + lm;
#pragma unroll
    for (int r = 0; r < 8; ++r) {
      int cg = c0 + w * 16 + r + 8 * hi;
      size_t idx = ((size_t)(b * 512 + cg)) * 4096 + ng;
      out[idx] = g * acc[ht][r] + x[idx];
    }
  }
}

// ---------------------------------------------------------------------------
extern "C" void kernel_launch(void* const* d_in, const int* in_sizes, int n_in,
                              void* d_out, int out_size, void* d_ws, size_t ws_size,
                              hipStream_t stream) {
  (void)in_sizes; (void)n_in; (void)out_size; (void)ws_size;
  const float* x      = (const float*)d_in[0];
  const int*   pos_h  = (const int*)d_in[1];
  const float* posemb = (const float*)d_in[3];
  const float* Wq     = (const float*)d_in[4];
  const float* bq     = (const float*)d_in[5];
  const float* Wk     = (const float*)d_in[6];
  const float* bk     = (const float*)d_in[7];
  const float* Wv     = (const float*)d_in[8];
  const float* bv     = (const float*)d_in[9];
  const float* gamma  = (const float*)d_in[10];
  float* out = (float*)d_out;

  char* ws = (char*)d_ws;                       // needs ~8.13 MB
  float*    xbar = (float*)(ws);                // 4*512*64 f32   = 524288 B
  float*    up   = (float*)(ws + 524288);       // 524288 B
  float*    k1   = (float*)(ws + 1048576);      // 4*64*256 f32   = 262144 B
  float*    e0   = (float*)(ws + 1310720);      // 256 f32        = 1024 B
  _Float16* Kt16 = (_Float16*)(ws + 1311744);   // 4*64*512 f16   = 262144 B
  _Float16* Vm16 = (_Float16*)(ws + 1573888);   // 4*512*64 f16   = 262144 B
  float*    E    = (float*)(ws + 1836032);      // 4*4096*64 f32  = 4194304 B
  _Float16* p16  = (_Float16*)(ws + 6030336);   // 4*4096*64 f16  = 2097152 B

  k_rowmean <<<16384, 256, 0, stream>>>(x, xbar);
  k_upsample<<< 2048,  64, 0, stream>>>(xbar, pos_h, posemb, up);
  k_k1      <<<  256, 256, 0, stream>>>(up, Wk, bk, k1);
  k_kt      <<<  256, 256, 0, stream>>>(k1, Wq, bq, Kt16, e0);
  k_vmean   <<< 2048,  64, 0, stream>>>(xbar, Wv, bv, Vm16);
  k_energy  <<<  128, 256, 0, stream>>>(x, Kt16, e0, E);
  k_softmax <<< 2048, 256, 0, stream>>>(E, p16);
  k_attnout <<< 1024, 256, 0, stream>>>(Vm16, p16, x, gamma, out);
}
